// NystromAttention_37735582662899
// MI455X (gfx1250) — matrix operand order repaired and checked by reading, not compiled, not run on hardware
//
#include <hip/hip_runtime.h>

#define BATCH 4
#define SEQ   4096
#define DIM   512
#define HEADS 8
#define DHEAD 64
#define NLM   256
#define BH    (BATCH*HEADS)   /* 32 */
#define BN    (BATCH*SEQ)     /* 16384 */
#define LGRP  (SEQ/NLM)       /* 16 */
#define KERN  33
#define QSCALE 0.125f         /* 64^-0.5 */

typedef __attribute__((ext_vector_type(16))) __bf16 v16bf;
typedef __attribute__((ext_vector_type(8)))  float  v8f;

union ABu { v16bf bf; unsigned int u[8]; uint4 q[2]; };

__device__ __forceinline__ unsigned short f2bf(float f) {
    unsigned u = __float_as_uint(f);
    return (unsigned short)((u + 0x7FFFu + ((u >> 16) & 1u)) >> 16);
}
__device__ __forceinline__ float bf2f(unsigned short u) {
    return __uint_as_float(((unsigned)u) << 16);
}

// 64x32 output block per wave: 4x2 grid of 16x16 WMMA tiles (~140 VGPRs, no spill).
// A bf16 row-major [M x K]; Bt = B^T bf16 row-major [N x K].
// Per 32-K step: 12 b128 loads feed 8 v_wmma ops (1.5 loads per WMMA).
// Fragment layouts per CDNA5 ISA 7.12.2 for v_wmma_f32_16x16x32_bf16.
__device__ __forceinline__ void wave_tile(const unsigned short* __restrict__ A, int lda,
                                          const unsigned short* __restrict__ Bt, int ldb,
                                          int K, int lane, v8f acc[4][2]) {
    const int rc   = lane & 15;
    const int half = lane >> 4;
    const unsigned short* pa = A  + (long)rc * lda + half * 8;
    const unsigned short* pb = Bt + (long)rc * ldb + half * 8;
    for (int k0 = 0; k0 < K; k0 += 32) {
        ABu af[4], bfr[2];
#pragma unroll
        for (int i = 0; i < 4; ++i) {
            const unsigned short* qa = pa + (long)i * 16 * lda;
            af[i].q[0] = *(const uint4*)(qa);        // K: half*8    ..+7
            af[i].q[1] = *(const uint4*)(qa + 16);   // K: 16+half*8 ..+7
        }
#pragma unroll
        for (int j = 0; j < 2; ++j) {
            const unsigned short* qb = pb + (long)j * 16 * ldb;
            bfr[j].q[0] = *(const uint4*)(qb);       // K: half*8    ..+7
            bfr[j].q[1] = *(const uint4*)(qb + 16);  // K: 16+half*8 ..+7
        }
#pragma unroll
        for (int i = 0; i < 4; ++i)
#pragma unroll
            for (int j = 0; j < 2; ++j)
                acc[i][j] = __builtin_amdgcn_wmma_f32_16x16x32_bf16(
                    false, af[i].bf, false, bfr[j].bf, (short)0, acc[i][j], false, false);
        pa += 32; pb += 32;
    }
}

#define ZERO_ACC(acc)                                                         \
    _Pragma("unroll") for (int i = 0; i < 4; ++i)                             \
    _Pragma("unroll") for (int j = 0; j < 2; ++j)                             \
    _Pragma("unroll") for (int r = 0; r < 8; ++r) acc[i][j][r] = 0.f;

// ---------------- generic batched WMMA GEMM (64x32 per wave) -------------
__global__ void __launch_bounds__(256, 1)
gemm_bf16(const unsigned short* __restrict__ A, long long sA, int lda,
          const unsigned short* __restrict__ Bt, long long sB, int ldb,
          float* __restrict__ C, unsigned short* __restrict__ Cbf,
          long long sC, int ldc,
          int M, int N, int K, float alpha,
          const float* __restrict__ bias) {
    const int lane = threadIdx.x & 31;
    const int wave = threadIdx.x >> 5;
    const int tn = N >> 5;
    const int ntiles = (M >> 6) * tn;
    const int tile = blockIdx.x * (blockDim.x >> 5) + wave;
    if (tile >= ntiles) return;
    const int tr = tile / tn, tc = tile % tn;
    const unsigned short* Ab = A  + (long long)blockIdx.z * sA + (long)(tr * 64) * lda;
    const unsigned short* Bb = Bt + (long long)blockIdx.z * sB + (long)(tc * 32) * ldb;
    v8f acc[4][2];
    ZERO_ACC(acc);
    wave_tile(Ab, lda, Bb, ldb, K, lane, acc);
    const int lc = lane & 15, lr = (lane >> 4) << 3;
    float*          Cf = C   ? C   + (long long)blockIdx.z * sC : nullptr;
    unsigned short* Cb = Cbf ? Cbf + (long long)blockIdx.z * sC : nullptr;
#pragma unroll
    for (int j = 0; j < 2; ++j) {
        const int col = tc * 32 + j * 16 + lc;
        const float bv = bias ? bias[col] : 0.f;
#pragma unroll
        for (int i = 0; i < 4; ++i) {
#pragma unroll
            for (int r = 0; r < 8; ++r) {
                const float val = alpha * acc[i][j][r] + bv;
                const long idx = (long)(tr * 64 + i * 16 + lr + r) * ldc + col;
                if (Cf) Cf[idx] = val;
                if (Cb) Cb[idx] = f2bf(val);
            }
        }
    }
}

// ---------------- QKV GEMM with head-split epilogue ----------------------
// A 32-wide column block lies inside exactly one head of one of q/k/v.
__global__ void __launch_bounds__(256, 1)
gemm_qkv(const unsigned short* __restrict__ Xbf,
         const unsigned short* __restrict__ Wt,
         unsigned short* __restrict__ Qbf,
         unsigned short* __restrict__ Kbf,
         float* __restrict__ V) {
    const int lane = threadIdx.x & 31;
    const int wave = threadIdx.x >> 5;
    const int tn = (3 * DIM) >> 5;                 // 48
    const int tile = blockIdx.x * (blockDim.x >> 5) + wave;
    if (tile >= (BN >> 6) * tn) return;
    const int tr = tile / tn, tc = tile % tn;
    v8f acc[4][2];
    ZERO_ACC(acc);
    wave_tile(Xbf + (long)(tr * 64) * DIM, DIM,
              Wt  + (long)(tc * 32) * DIM, DIM, DIM, lane, acc);
    const int lc = lane & 15, lr = (lane >> 4) << 3;
    const int col0 = tc * 32;
    const int sec = col0 >> 9;            // 0=q 1=k 2=v
    const int h   = (col0 >> 6) & 7;
    const int dh0 = col0 & 63;
#pragma unroll
    for (int j = 0; j < 2; ++j) {
        const int dh = dh0 + j * 16 + lc;
#pragma unroll
        for (int i = 0; i < 4; ++i) {
#pragma unroll
            for (int r = 0; r < 8; ++r) {
                const int row = tr * 64 + i * 16 + lr + r;
                const int b = row >> 12, sq = row & 4095;
                const long di = (((long)(b * HEADS + h) * SEQ + sq) * DHEAD + dh);
                const float v = acc[i][j][r];
                if (sec == 0)      Qbf[di] = f2bf(v * QSCALE);
                else if (sec == 1) Kbf[di] = f2bf(v);
                else               V[di] = v;
            }
        }
    }
}

// ---------------- f32 -> bf16 elementwise --------------------------------
__global__ void to_bf16(const float* __restrict__ in, unsigned short* __restrict__ out, int n) {
    const int idx = blockIdx.x * blockDim.x + threadIdx.x;
    if (idx < n) out[idx] = f2bf(in[idx]);
}

// ---------------- transpose (f32 -> bf16), fused  diag*I + scale*X^T -----
__global__ void transpose_bf(const float* __restrict__ in, unsigned short* __restrict__ out,
                             int R, int C, float scale, float diag) {
    const long base = (long)blockIdx.z * R * C;
    const int idx = blockIdx.x * blockDim.x + threadIdx.x;
    if (idx >= R * C) return;
    const int r = idx / C, c = idx % C;
    float v = scale * in[base + idx];
    if (r == c) v += diag;
    out[base + (long)c * R + r] = f2bf(v);
}

// ---------------- landmark means (bf16 in, bf16 out, f32 acc) ------------
__global__ void landmarks(const unsigned short* __restrict__ Qbf,
                          const unsigned short* __restrict__ Kbf,
                          unsigned short* __restrict__ QLbf,
                          unsigned short* __restrict__ KLbf) {
    const int idx = blockIdx.x * blockDim.x + threadIdx.x;
    if (idx >= BH * NLM * DHEAD) return;
    const int dh = idx % DHEAD, mi = (idx / DHEAD) % NLM, bh = idx / (NLM * DHEAD);
    const long base = (long)bh * SEQ * DHEAD + (long)(mi * LGRP) * DHEAD + dh;
    float qs = 0.f, ks = 0.f;
#pragma unroll
    for (int j = 0; j < LGRP; ++j) { qs += bf2f(Qbf[base + j * DHEAD]); ks += bf2f(Kbf[base + j * DHEAD]); }
    QLbf[idx] = f2bf(qs * (1.f / LGRP));
    KLbf[idx] = f2bf(ks * (1.f / LGRP));
}

// ---------------- row softmax, f32 in place ------------------------------
__global__ void softmax_rows(float* __restrict__ d, int cols) {
    __shared__ float red[256];
    const long base = (long)blockIdx.x * cols;
    float m = -3.4e38f;
    for (int c = threadIdx.x; c < cols; c += 256) m = fmaxf(m, d[base + c]);
    red[threadIdx.x] = m; __syncthreads();
    for (int s = 128; s; s >>= 1) { if (threadIdx.x < s) red[threadIdx.x] = fmaxf(red[threadIdx.x], red[threadIdx.x + s]); __syncthreads(); }
    m = red[0]; __syncthreads();
    float sum = 0.f;
    for (int c = threadIdx.x; c < cols; c += 256) { float e = __expf(d[base + c] - m); d[base + c] = e; sum += e; }
    red[threadIdx.x] = sum; __syncthreads();
    for (int s = 128; s; s >>= 1) { if (threadIdx.x < s) red[threadIdx.x] += red[threadIdx.x + s]; __syncthreads(); }
    const float inv = 1.f / red[0];
    for (int c = threadIdx.x; c < cols; c += 256) d[base + c] *= inv;
}

// ---------------- row softmax, bf16 in place (f32 internal) --------------
__global__ void softmax_rows_bf(unsigned short* __restrict__ d, int cols) {
    __shared__ float red[256];
    const long base = (long)blockIdx.x * cols;
    float m = -3.4e38f;
    for (int c = threadIdx.x; c < cols; c += 256) m = fmaxf(m, bf2f(d[base + c]));
    red[threadIdx.x] = m; __syncthreads();
    for (int s = 128; s; s >>= 1) { if (threadIdx.x < s) red[threadIdx.x] = fmaxf(red[threadIdx.x], red[threadIdx.x + s]); __syncthreads(); }
    m = red[0]; __syncthreads();
    float sum = 0.f;
    for (int c = threadIdx.x; c < cols; c += 256) { float e = __expf(bf2f(d[base + c]) - m); d[base + c] = f2bf(e); sum += e; }
    red[threadIdx.x] = sum; __syncthreads();
    for (int s = 128; s; s >>= 1) { if (threadIdx.x < s) red[threadIdx.x] += red[threadIdx.x + s]; __syncthreads(); }
    const float inv = 1.f / red[0];
    for (int c = threadIdx.x; c < cols; c += 256) d[base + c] = f2bf(bf2f(d[base + c]) * inv);
}

// ---------------- pinv scaling helpers -----------------------------------
__global__ void zero2(unsigned int* s) { if (threadIdx.x < 2) s[threadIdx.x] = 0u; }

__global__ void pinv_sums(const float* __restrict__ x, unsigned int* __restrict__ s) {
    const int t = blockIdx.x * blockDim.x + threadIdx.x;
    if (t >= 2 * BH * NLM) return;
    const int half = t / (BH * NLM), r = t % (BH * NLM);
    const int bh = r / NLM, i = r % NLM;
    const float* xb = x + (long)bh * NLM * NLM;
    float sum = 0.f;
    if (half == 0) for (int j = 0; j < NLM; ++j) sum += fabsf(xb[i * NLM + j]);
    else           for (int j = 0; j < NLM; ++j) sum += fabsf(xb[j * NLM + i]);
    atomicMax(&s[half], __float_as_uint(sum));
}

__global__ void pinv_init_z(const float* __restrict__ x, const unsigned int* __restrict__ s,
                            float* __restrict__ z, unsigned short* __restrict__ zbf) {
    const int idx = blockIdx.x * blockDim.x + threadIdx.x;
    if (idx >= BH * NLM * NLM) return;
    const float inv = 1.f / (__uint_as_float(s[0]) * __uint_as_float(s[1]));
    const int j = idx % NLM, i = (idx / NLM) % NLM, bh = idx / (NLM * NLM);
    const float v = x[(long)bh * NLM * NLM + j * NLM + i] * inv;
    z[(long)bh * NLM * NLM + i * NLM + j]   = v;
    zbf[(long)bh * NLM * NLM + i * NLM + j] = f2bf(v);
}

// ---------------- conv residual + head merge -> bf16 ---------------------
__global__ void fuse_conv(const float* __restrict__ oa, const float* __restrict__ V,
                          const float* __restrict__ cw, unsigned short* __restrict__ ybf) {
    const int idx = blockIdx.x * blockDim.x + threadIdx.x;
    if (idx >= BH * SEQ * DHEAD) return;
    const int dh = idx & 63, i = (idx >> 6) & 4095, bh = idx >> 18;
    const int h = bh & (HEADS - 1), b = bh >> 3;
    float acc = oa[idx];
    const float* vb = V + (long)bh * SEQ * DHEAD + dh;
#pragma unroll
    for (int t = 0; t < KERN; ++t) {
        const int s = i + t - (KERN / 2);
        if (s >= 0 && s < SEQ) acc += vb[(long)s * DHEAD] * cw[h * KERN + t];
    }
    ybf[((long)(b * SEQ + i)) * (HEADS * DHEAD) + h * DHEAD + dh] = f2bf(acc);
}

// =========================================================================
extern "C" void kernel_launch(void* const* d_in, const int* in_sizes, int n_in,
                              void* d_out, int out_size, void* d_ws, size_t ws_size,
                              hipStream_t stream) {
    const float* x      = (const float*)d_in[0];
    const float* w_qkv  = (const float*)d_in[1];
    const float* w_out  = (const float*)d_in[2];
    const float* b_out  = (const float*)d_in[3];
    const float* conv_w = (const float*)d_in[4];
    float* out = (float*)d_out;

    char* ws = (char*)d_ws;
    size_t off = 0;
    auto alloc = [&](size_t bytes) -> void* {
        void* p = ws + off; off = (off + bytes + 255) & ~(size_t)255; return p;
    };

    unsigned short* xbf   = (unsigned short*)alloc((size_t)BN * DIM * 2);
    unsigned short* qbf   = (unsigned short*)alloc((size_t)BH * SEQ * DHEAD * 2);
    unsigned short* kbf   = (unsigned short*)alloc((size_t)BH * SEQ * DHEAD * 2);
    float*          vf    = (float*)alloc((size_t)BH * SEQ * DHEAD * 4);
    unsigned short* vT    = (unsigned short*)alloc((size_t)BH * SEQ * DHEAD * 2);
    unsigned short* qlbf  = (unsigned short*)alloc((size_t)BH * NLM * DHEAD * 2);
    unsigned short* klbf  = (unsigned short*)alloc((size_t)BH * NLM * DHEAD * 2);
    unsigned short* sim1b = (unsigned short*)alloc((size_t)BH * SEQ * NLM * 2);
    unsigned short* sim3b = (unsigned short*)alloc((size_t)BH * NLM * SEQ * 2);
    float*          x2    = (float*)alloc((size_t)BH * NLM * NLM * 4);
    unsigned short* x2bf  = (unsigned short*)alloc((size_t)BH * NLM * NLM * 2);
    float*          zA    = (float*)alloc((size_t)BH * NLM * NLM * 4);
    unsigned short* zAbf  = (unsigned short*)alloc((size_t)BH * NLM * NLM * 2);
    float*          zB    = (float*)alloc((size_t)BH * NLM * NLM * 4);
    unsigned short* zBbf  = (unsigned short*)alloc((size_t)BH * NLM * NLM * 2);
    float*          xz    = (float*)alloc((size_t)BH * NLM * NLM * 4);
    unsigned short* xzbf  = (unsigned short*)alloc((size_t)BH * NLM * NLM * 2);
    float*          tmp   = (float*)alloc((size_t)BH * NLM * NLM * 4);
    unsigned short* tbf   = (unsigned short*)alloc((size_t)BH * NLM * NLM * 2);
    float*          a3v   = (float*)alloc((size_t)BH * NLM * DHEAD * 4);
    unsigned short* a3vT  = (unsigned short*)alloc((size_t)BH * NLM * DHEAD * 2);
    float*          w2    = (float*)alloc((size_t)BH * NLM * DHEAD * 4);
    unsigned short* w2T   = (unsigned short*)alloc((size_t)BH * NLM * DHEAD * 2);
    unsigned short* wqkvT = (unsigned short*)alloc((size_t)DIM * 3 * DIM * 2);
    unsigned short* woutT = (unsigned short*)alloc((size_t)DIM * DIM * 2);
    unsigned int*   scal  = (unsigned int*)alloc(256);
    float*          oa  = (float*)sim3b;           // alias: attn3 bf consumed first
    unsigned short* ybf = (unsigned short*)sim1b;  // alias: attn1 bf consumed first

    const dim3 B256(256);
    const long long sQK = (long long)SEQ * DHEAD;
    const long long sL  = (long long)NLM * DHEAD;
    const long long sMM = (long long)NLM * NLM;
    auto blocks = [](int m, int n) { return dim3((((m >> 6) * (n >> 5)) + 7) / 8); };
    auto blocksz = [](int m, int n, int z) { return dim3((((m >> 6) * (n >> 5)) + 7) / 8, 1, z); };

    // 1-2. input + weight conversions
    to_bf16<<<dim3((BN * DIM) / 256), B256, 0, stream>>>(x, xbf, BN * DIM);
    transpose_bf<<<dim3((DIM * 3 * DIM + 255) / 256), B256, 0, stream>>>(w_qkv, wqkvT, DIM, 3 * DIM, 1.f, 0.f);
    // 3. QKV GEMM
    gemm_qkv<<<blocks(BN, 3 * DIM), B256, 0, stream>>>(xbf, wqkvT, qbf, kbf, vf);
    // 4. landmark means
    landmarks<<<dim3((BH * NLM * DHEAD) / 256), B256, 0, stream>>>(qbf, kbf, qlbf, klbf);
    // 5. sim1 = q @ k_l^T -> bf16
    gemm_bf16<<<blocksz(SEQ, NLM, BH), B256, 0, stream>>>(qbf, sQK, DHEAD, klbf, sL, DHEAD,
        nullptr, sim1b, (long long)SEQ * NLM, NLM, SEQ, NLM, DHEAD, 1.f, nullptr);
    // 6. sim2 = q_l @ k_l^T -> f32
    gemm_bf16<<<blocksz(NLM, NLM, BH), B256, 0, stream>>>(qlbf, sL, DHEAD, klbf, sL, DHEAD,
        x2, nullptr, sMM, NLM, NLM, NLM, DHEAD, 1.f, nullptr);
    // 7. sim3 = q_l @ k^T -> bf16
    gemm_bf16<<<blocksz(NLM, SEQ, BH), B256, 0, stream>>>(qlbf, sL, DHEAD, kbf, sQK, DHEAD,
        nullptr, sim3b, (long long)NLM * SEQ, SEQ, NLM, SEQ, DHEAD, 1.f, nullptr);
    // 8-10. softmaxes
    softmax_rows_bf<<<dim3(BH * SEQ), B256, 0, stream>>>(sim1b, NLM);
    softmax_rows<<<dim3(BH * NLM), B256, 0, stream>>>(x2, NLM);
    softmax_rows_bf<<<dim3(BH * NLM), B256, 0, stream>>>(sim3b, SEQ);
    // 11. attn2 bf16 copy for WMMA A-side
    to_bf16<<<dim3((BH * NLM * NLM) / 256), B256, 0, stream>>>(x2, x2bf, BH * NLM * NLM);
    // 12. pinv init: z = x^T / (max rowsum * max colsum)
    zero2<<<dim3(1), dim3(32), 0, stream>>>(scal);
    pinv_sums<<<dim3((2 * BH * NLM) / 256), B256, 0, stream>>>(x2, scal);
    pinv_init_z<<<dim3((BH * NLM * NLM) / 256), B256, 0, stream>>>(x2, scal, zA, zAbf);
    // 13. Newton-Schulz iterations
    float *z = zA, *zn = zB;
    unsigned short *zbf = zAbf, *znbf = zBbf;
    const dim3 gT(dim3((NLM * NLM + 255) / 256, 1, BH));
    const dim3 gG = blocksz(NLM, NLM, BH);
    for (int it = 0; it < 6; ++it) {
        transpose_bf<<<gT, B256, 0, stream>>>(z, tbf, NLM, NLM, 1.f, 0.f);             // z^T
        gemm_bf16<<<gG, B256, 0, stream>>>(x2bf, sMM, NLM, tbf, sMM, NLM,
            xz, xzbf, sMM, NLM, NLM, NLM, NLM, 1.f, nullptr);                          // xz = x@z
        transpose_bf<<<gT, B256, 0, stream>>>(xz, tbf, NLM, NLM, -1.f, 7.f);           // (7I-xz)^T
        gemm_bf16<<<gG, B256, 0, stream>>>(xzbf, sMM, NLM, tbf, sMM, NLM,
            tmp, nullptr, sMM, NLM, NLM, NLM, NLM, 1.f, nullptr);                      // xz@(7I-xz)
        transpose_bf<<<gT, B256, 0, stream>>>(tmp, tbf, NLM, NLM, -1.f, 15.f);         // (15I-..)^T
        gemm_bf16<<<gG, B256, 0, stream>>>(xzbf, sMM, NLM, tbf, sMM, NLM,
            tmp, nullptr, sMM, NLM, NLM, NLM, NLM, 1.f, nullptr);                      // xz@(15I-..)
        transpose_bf<<<gT, B256, 0, stream>>>(tmp, tbf, NLM, NLM, -1.f, 13.f);         // (13I-..)^T
        gemm_bf16<<<gG, B256, 0, stream>>>(zbf, sMM, NLM, tbf, sMM, NLM,
            zn, znbf, sMM, NLM, NLM, NLM, NLM, 0.25f, nullptr);                        // 0.25 z@..
        float* tf = z; z = zn; zn = tf;
        unsigned short* tb = zbf; zbf = znbf; znbf = tb;
    }
    // 14. v^T -> bf16
    transpose_bf<<<dim3((SEQ * DHEAD + 255) / 256, 1, BH), B256, 0, stream>>>(vf, vT, SEQ, DHEAD, 1.f, 0.f);
    // 15. a3v = attn3 @ v
    gemm_bf16<<<blocksz(NLM, DHEAD, BH), B256, 0, stream>>>(sim3b, (long long)NLM * SEQ, SEQ,
        vT, sQK, SEQ, a3v, nullptr, sL, DHEAD, NLM, DHEAD, SEQ, 1.f, nullptr);
    // 16-17. w2 = attn2_inv @ a3v
    transpose_bf<<<dim3((NLM * DHEAD + 255) / 256, 1, BH), B256, 0, stream>>>(a3v, a3vT, NLM, DHEAD, 1.f, 0.f);
    gemm_bf16<<<blocksz(NLM, DHEAD, BH), B256, 0, stream>>>(zbf, sMM, NLM,
        a3vT, sL, NLM, w2, nullptr, sL, DHEAD, NLM, DHEAD, NLM, 1.f, nullptr);
    // 18-19. out_attn = attn1 @ w2  (writes into sim3b storage as f32)
    transpose_bf<<<dim3((NLM * DHEAD + 255) / 256, 1, BH), B256, 0, stream>>>(w2, w2T, NLM, DHEAD, 1.f, 0.f);
    gemm_bf16<<<blocksz(SEQ, DHEAD, BH), B256, 0, stream>>>(sim1b, (long long)SEQ * NLM, NLM,
        w2T, sL, NLM, oa, nullptr, sQK, DHEAD, SEQ, DHEAD, NLM, 1.f, nullptr);
    // 20. + depthwise conv residual, merge heads -> bf16 (into sim1b storage)
    fuse_conv<<<dim3((BH * SEQ * DHEAD) / 256), B256, 0, stream>>>(oa, vf, conv_w, ybf);
    // 21-22. final projection with bias
    transpose_bf<<<dim3((DIM * DIM + 255) / 256), B256, 0, stream>>>(w_out, woutT, DIM, DIM, 1.f, 0.f);
    gemm_bf16<<<blocks(BN, DIM), B256, 0, stream>>>(ybf, 0, DIM,
        woutT, 0, DIM, out, nullptr, 0, DIM, BN, DIM, DIM, 1.f, b_out);

    (void)in_sizes; (void)n_in; (void)out_size; (void)ws_size;
}
